// GiantGraphMPNN_45586782880046
// MI455X (gfx1250) — compile-verified, matrix operand order checked by reading, not run on hardware
//
#include <hip/hip_runtime.h>

typedef __attribute__((ext_vector_type(16))) _Float16 v16h;
typedef __attribute__((ext_vector_type(8)))  _Float16 v8h;
typedef __attribute__((ext_vector_type(8)))  float    v8f;
typedef __attribute__((ext_vector_type(4)))  float    v4f;

#define IN_CH 256
#define HID   16
#define NW    64   // p2p|d2p|p2d|sl concatenated along N

// ---------------- utility kernels ----------------

__global__ void k_zero(float* __restrict__ p, int n) {
  int i = blockIdx.x * blockDim.x + threadIdx.x;
  if (i < n) p[i] = 0.0f;
}

__global__ void k_degree(const int* __restrict__ col, float* __restrict__ deg, int n_edges) {
  int e = blockIdx.x * blockDim.x + threadIdx.x;
  if (e < n_edges) atomicAdd(&deg[col[e]], 1.0f);
}

__global__ void k_dinv_isd(const float* __restrict__ x, const float* __restrict__ deg,
                           float* __restrict__ dinv, float* __restrict__ isd, int n) {
  int i = blockIdx.x * blockDim.x + threadIdx.x;
  if (i < n) {
    dinv[i] = rsqrtf(deg[i]);
    isd[i]  = x[(size_t)i * IN_CH];   // is_drug flag lives in channel 0
  }
}

// Transpose+convert the 4 layer-1 weights into wT1[64][256] f16 (N-major),
// the 4 layer-2 weights into wT2[64][32] f16 (K zero-padded 16->32),
// and concatenate biases.
__global__ void k_prep(const float* __restrict__ w1a, const float* __restrict__ b1a,
                       const float* __restrict__ w1b, const float* __restrict__ b1b,
                       const float* __restrict__ w1c, const float* __restrict__ b1c,
                       const float* __restrict__ w1d, const float* __restrict__ b1d,
                       const float* __restrict__ w2a, const float* __restrict__ b2a,
                       const float* __restrict__ w2b, const float* __restrict__ b2b,
                       const float* __restrict__ w2c, const float* __restrict__ b2c,
                       const float* __restrict__ w2d, const float* __restrict__ b2d,
                       _Float16* __restrict__ wT1, _Float16* __restrict__ wT2,
                       float* __restrict__ bc1, float* __restrict__ bc2) {
  int t = blockIdx.x * blockDim.x + threadIdx.x;   // 0..16383
  if (t < NW * IN_CH) {
    int r = t >> 8, k = t & 255;        // r = output col (0..63), k = input channel
    int g = r >> 4, n = r & 15;
    const float* w1 = (g == 0) ? w1a : (g == 1) ? w1b : (g == 2) ? w1c : w1d;
    wT1[(size_t)r * IN_CH + k] = (_Float16)w1[k * HID + n];
  }
  if (t < NW * 32) {
    int r2 = t >> 5, k2 = t & 31;
    int g2 = r2 >> 4, n2 = r2 & 15;
    const float* w2 = (g2 == 0) ? w2a : (g2 == 1) ? w2b : (g2 == 2) ? w2c : w2d;
    wT2[(size_t)r2 * 32 + k2] = (k2 < HID) ? (_Float16)w2[k2 * HID + n2] : (_Float16)0.0f;
  }
  if (t < NW) {
    int g3 = t >> 4, n3 = t & 15;
    const float* bb1 = (g3 == 0) ? b1a : (g3 == 1) ? b1b : (g3 == 2) ? b1c : b1d;
    const float* bb2 = (g3 == 0) ? b2a : (g3 == 1) ? b2b : (g3 == 2) ? b2c : b2d;
    bc1[t] = bb1[n3];
    bc2[t] = bb2[n3];
  }
}

// ---------------- WMMA GEMM kernels ----------------
// Block = 256 threads = 8 waves. Each wave owns a distinct 16-row M-tile and
// computes ALL four 16-wide N-tiles of hpre[N][64]: one A fetch+f16 convert
// feeds 4 WMMAs per K-step (no redundant A traffic on the 102MB x stream).
// A fragment (16x32 f16): lane m = lane&15, K-half = (lane>>4)*8,
//   v16h elems 0..7 = K kb..kb+7, elems 8..15 = K kb+16..kb+23.
// B fragments come from N-major (transposed) f16 weights => identical
// per-lane indexing with lane acting as the N column.

__global__ void __launch_bounds__(256)
k_gemm1(const float* __restrict__ x, const _Float16* __restrict__ wT1,
        const float* __restrict__ bc1, float* __restrict__ hpre, int n_nodes) {
  const int wave  = threadIdx.x >> 5;                 // 0..7
  const int lane  = threadIdx.x & 31;
  const int m0    = (blockIdx.x * 8 + wave) * 16;
  if (m0 >= n_nodes) return;                          // wave-uniform: EXEC stays full
  const int nlo   = lane & 15;
  const int khalf = (lane >> 4) << 3;                 // 0 or 8

  int row = m0 + nlo;
  if (row >= n_nodes) row = n_nodes - 1;              // safe clamp (100000 % 16 == 0)

  v8f c[4];
#pragma unroll
  for (int t = 0; t < 4; ++t) {
    float bv = bc1[t * 16 + nlo];                     // bias broadcast: C[m][n] = b[n]
#pragma unroll
    for (int i = 0; i < 8; ++i) c[t][i] = bv;
  }

  const float*    xrow = x   + (size_t)row * IN_CH;
  const _Float16* w0   = wT1 + (size_t)nlo * IN_CH;   // tile t adds t*16 rows

#pragma unroll
  for (int kk = 0; kk < IN_CH; kk += 32) {
    const int kb = kk + khalf;
    v4f f0 = *(const v4f*)(xrow + kb);
    v4f f1 = *(const v4f*)(xrow + kb + 4);
    v4f f2 = *(const v4f*)(xrow + kb + 16);
    v4f f3 = *(const v4f*)(xrow + kb + 20);
    v16h a;
#pragma unroll
    for (int j = 0; j < 4; ++j) {
      a[j]      = (_Float16)f0[j];
      a[4 + j]  = (_Float16)f1[j];
      a[8 + j]  = (_Float16)f2[j];
      a[12 + j] = (_Float16)f3[j];
    }
#pragma unroll
    for (int t = 0; t < 4; ++t) {
      const _Float16* wrow = w0 + (size_t)(t * 16) * IN_CH;
      v8h b0 = *(const v8h*)(wrow + kb);
      v8h b1 = *(const v8h*)(wrow + kb + 16);
      v16h b;
#pragma unroll
      for (int j = 0; j < 8; ++j) { b[j] = b0[j]; b[8 + j] = b1[j]; }
      c[t] = __builtin_amdgcn_wmma_f32_16x16x32_f16(false, a, false, b,
                                                    (short)0, c[t], false, false);
    }
  }

  const int mbase = m0 + ((lane >> 4) << 3);
#pragma unroll
  for (int t = 0; t < 4; ++t) {
#pragma unroll
    for (int r = 0; r < 8; ++r) {
      int rr = mbase + r;
      if (rr < n_nodes) hpre[(size_t)rr * NW + t * 16 + nlo] = c[t][r];
    }
  }
}

__global__ void __launch_bounds__(256)
k_gemm2(const float* __restrict__ h, const _Float16* __restrict__ wT2,
        const float* __restrict__ bc2, float* __restrict__ hpre, int n_nodes) {
  const int wave  = threadIdx.x >> 5;
  const int lane  = threadIdx.x & 31;
  const int m0    = (blockIdx.x * 8 + wave) * 16;
  if (m0 >= n_nodes) return;
  const int nlo   = lane & 15;
  const int khalf = (lane >> 4) << 3;

  int row = m0 + nlo;
  if (row >= n_nodes) row = n_nodes - 1;

  // A: K=16 real, padded to 32 with zeros (elems 8..15 = K 16..31 region = 0)
  v4f g0 = *(const v4f*)(h + (size_t)row * HID + khalf);
  v4f g1 = *(const v4f*)(h + (size_t)row * HID + khalf + 4);
  v16h a;
#pragma unroll
  for (int j = 0; j < 4; ++j) {
    a[j]      = (_Float16)g0[j];
    a[4 + j]  = (_Float16)g1[j];
    a[8 + j]  = (_Float16)0.0f;
    a[12 + j] = (_Float16)0.0f;
  }

  v8f c[4];
#pragma unroll
  for (int t = 0; t < 4; ++t) {
    float bv = bc2[t * 16 + nlo];
#pragma unroll
    for (int i = 0; i < 8; ++i) c[t][i] = bv;
  }

  const _Float16* w0 = wT2 + (size_t)nlo * 32;
#pragma unroll
  for (int t = 0; t < 4; ++t) {
    const _Float16* wrow = w0 + (size_t)(t * 16) * 32;
    v8h b0 = *(const v8h*)(wrow + khalf);        // K kb..kb+7 (real)
    v8h b1 = *(const v8h*)(wrow + khalf + 16);   // K kb+16..kb+23 (zero-padded)
    v16h b;
#pragma unroll
    for (int j = 0; j < 8; ++j) { b[j] = b0[j]; b[8 + j] = b1[j]; }
    c[t] = __builtin_amdgcn_wmma_f32_16x16x32_f16(false, a, false, b,
                                                  (short)0, c[t], false, false);
  }

  const int mbase = m0 + ((lane >> 4) << 3);
#pragma unroll
  for (int t = 0; t < 4; ++t) {
#pragma unroll
    for (int r = 0; r < 8; ++r) {
      int rr = mbase + r;
      if (rr < n_nodes) hpre[(size_t)rr * NW + t * 16 + nlo] = c[t][r];
    }
  }
}

// ---------------- edge message pass ----------------
// lane t -> edge e = t/16, channel c = t%16. Gather of hpre row (64B) is
// coalesced across the 16 lanes; the 16 atomic f32 adds per edge hit
// contiguous addresses of the L2-resident 6.4MB agg buffer.
__global__ void k_edge(const int* __restrict__ row, const int* __restrict__ col,
                       const float* __restrict__ isd, const float* __restrict__ dinv,
                       const float* __restrict__ hpre, float* __restrict__ agg,
                       int n_edges) {
  long long t = (long long)blockIdx.x * blockDim.x + threadIdx.x;
  int e = (int)(t >> 4), c = (int)(t & 15);
  if (e < n_edges) {
    int r = row[e], d = col[e];
    float et = isd[r] + 2.0f * isd[d];          // exact {0,1,2,3}
    float nm = (et == 3.0f) ? 0.0f : dinv[r] * dinv[d];
    int sel = (et == 1.0f) ? 16 : ((et == 2.0f) ? 32 : 0);  // d2p / p2d / p2p
    float v = hpre[(size_t)r * NW + sel + c] * nm;
    atomicAdd(&agg[(size_t)d * HID + c], v);
  }
}

// h = (relu?) (agg + self-loop slice of hpre)
__global__ void k_combine(const float* __restrict__ agg, const float* __restrict__ hpre,
                          float* __restrict__ h, int count, int relu) {
  int t = blockIdx.x * blockDim.x + threadIdx.x;
  if (t < count) {
    int i = t >> 4, c = t & 15;
    float v = agg[t] + hpre[(size_t)i * NW + 48 + c];
    h[t] = relu ? fmaxf(v, 0.0f) : v;
  }
}

// out[b] = x1^T P x2
__global__ void k_pred(const float* __restrict__ h2, const int* __restrict__ dp,
                       const int* __restrict__ cell, const float* __restrict__ pred,
                       float* __restrict__ out, int batch) {
  int b = blockIdx.x * blockDim.x + threadIdx.x;
  if (b >= batch) return;
  const float* x1 = h2 + (size_t)dp[2 * b]     * HID;
  const float* x2 = h2 + (size_t)dp[2 * b + 1] * HID;
  const float* P  = pred + (size_t)cell[b] * HID * HID;
  float xa[HID], xb[HID];
#pragma unroll
  for (int i = 0; i < HID; ++i) { xa[i] = x1[i]; xb[i] = x2[i]; }
  float acc = 0.0f;
#pragma unroll
  for (int i = 0; i < HID; ++i) {
    float ti = 0.0f;
#pragma unroll
    for (int j = 0; j < HID; ++j) ti = fmaf(P[i * HID + j], xb[j], ti);
    acc = fmaf(xa[i], ti, acc);
  }
  out[b] = acc;
}

// ---------------- launch ----------------

extern "C" void kernel_launch(void* const* d_in, const int* in_sizes, int n_in,
                              void* d_out, int out_size, void* d_ws, size_t ws_size,
                              hipStream_t stream) {
  const float* x    = (const float*)d_in[0];
  const int*   ei   = (const int*)d_in[1];
  const int*   dp   = (const int*)d_in[2];
  const int*   cell = (const int*)d_in[3];
  const float* w1a = (const float*)d_in[4],  *b1a = (const float*)d_in[5];
  const float* w1b = (const float*)d_in[6],  *b1b = (const float*)d_in[7];
  const float* w1c = (const float*)d_in[8],  *b1c = (const float*)d_in[9];
  const float* w1d = (const float*)d_in[10], *b1d = (const float*)d_in[11];
  const float* w2a = (const float*)d_in[12], *b2a = (const float*)d_in[13];
  const float* w2b = (const float*)d_in[14], *b2b = (const float*)d_in[15];
  const float* w2c = (const float*)d_in[16], *b2c = (const float*)d_in[17];
  const float* w2d = (const float*)d_in[18], *b2d = (const float*)d_in[19];
  const float* pred = (const float*)d_in[20];
  float* out = (float*)d_out;

  const int n_nodes = in_sizes[0] / IN_CH;
  const int n_edges = in_sizes[1] / 2;
  const int batch   = in_sizes[3];
  const int* erow = ei;
  const int* ecol = ei + n_edges;

  // workspace carve-out (256B aligned)
  auto al = [](size_t v) { return (v + 255) & ~(size_t)255; };
  char* w = (char*)d_ws;
  size_t o = 0;
  float*    deg  = (float*)(w + o);    o += al((size_t)n_nodes * 4);
  float*    dinv = (float*)(w + o);    o += al((size_t)n_nodes * 4);
  float*    isd  = (float*)(w + o);    o += al((size_t)n_nodes * 4);
  _Float16* wT1  = (_Float16*)(w + o); o += al((size_t)NW * IN_CH * 2);
  _Float16* wT2  = (_Float16*)(w + o); o += al((size_t)NW * 32 * 2);
  float*    bc1  = (float*)(w + o);    o += 256;
  float*    bc2  = (float*)(w + o);    o += 256;
  float*    hpre = (float*)(w + o);    o += al((size_t)n_nodes * NW * 4);
  float*    agg  = (float*)(w + o);    o += al((size_t)n_nodes * HID * 4);
  float*    hbuf = (float*)(w + o);    o += al((size_t)n_nodes * HID * 4);
  (void)ws_size; (void)n_in; (void)out_size;

  const int B = 256;
  const int nC   = n_nodes * HID;
  const int gN   = (n_nodes + B - 1) / B;
  const int gC   = (nC + B - 1) / B;
  const int gE   = (n_edges + B - 1) / B;
  const int gE16 = (int)(((long long)n_edges * 16 + B - 1) / B);
  const int nTile = (n_nodes + 15) / 16;
  const int gM   = (nTile + 7) / 8;           // 8 waves/block, one 16-row tile per wave
  const int gB   = (batch + B - 1) / B;

  // degree + norm prep
  k_zero<<<gN, B, 0, stream>>>(deg, n_nodes);
  k_zero<<<gC, B, 0, stream>>>(agg, nC);
  k_degree<<<gE, B, 0, stream>>>(ecol, deg, n_edges);
  k_prep<<<(NW * IN_CH + B - 1) / B, B, 0, stream>>>(
      w1a, b1a, w1b, b1b, w1c, b1c, w1d, b1d,
      w2a, b2a, w2b, b2b, w2c, b2c, w2d, b2d, wT1, wT2, bc1, bc2);
  k_dinv_isd<<<gN, B, 0, stream>>>(x, deg, dinv, isd, n_nodes);

  // layer 1
  k_gemm1<<<gM, B, 0, stream>>>(x, wT1, bc1, hpre, n_nodes);
  k_edge<<<gE16, B, 0, stream>>>(erow, ecol, isd, dinv, hpre, agg, n_edges);
  k_combine<<<gC, B, 0, stream>>>(agg, hpre, hbuf, nC, 1);

  // layer 2
  k_zero<<<gC, B, 0, stream>>>(agg, nC);
  k_gemm2<<<gM, B, 0, stream>>>(hbuf, wT2, bc2, hpre, n_nodes);
  k_edge<<<gE16, B, 0, stream>>>(erow, ecol, isd, dinv, hpre, agg, n_edges);
  k_combine<<<gC, B, 0, stream>>>(agg, hpre, hbuf, nC, 0);

  // bilinear predictor
  k_pred<<<gB, B, 0, stream>>>(hbuf, dp, cell, pred, out, batch);
}
